// Model_64587718197884
// MI455X (gfx1250) — matrix operand order repaired and checked by reading, not compiled, not run on hardware
//
#include <hip/hip_runtime.h>

// ---------------------------------------------------------------------------
// Model dims
// ---------------------------------------------------------------------------
#define NN   250
#define BB   32
#define TT   12
#define WW   12
#define HEADS 5
#define DHEAD 50
#define TS5  60
#define OCT  240
#define H3   750          // 3*N
#define NBW  96000        // N*B*W
#define NB   8000         // N*B (also B*N rows)

#define ACT_NONE  0
#define ACT_RELU  1
#define ACT_SIG   2
#define ACT_LRELU 3

typedef __attribute__((ext_vector_type(16))) _Float16 v16h;
typedef __attribute__((ext_vector_type(8)))  _Float16 v8h;
typedef __attribute__((ext_vector_type(8)))  float    v8f;

__device__ __forceinline__ float sigm(float v) { return 1.f / (1.f + __expf(-v)); }

// ---------------------------------------------------------------------------
// Generic batched GEMM:  C[M,N] = act( A[M,K] @ Bw[N,K]^T (+bias) (+C if accum) )
// Row-major A (lda), Bw (ldb), C (ldc).  Batch z: z1=z/bmod, z2=z%bmod,
// per-matrix offsets z1*s?1 + z2*s?2.
// Block tile 64x64, BK=32, 8 waves; each wave owns two 16x16 WMMA tiles.
// LDS double-buffered; f32 -> f16 conversion during staging; f32 accumulate.
// Staging is branch-free (clamped loads) with a vectorized full-tile path:
// 4x float2 global loads -> 8x v_cvt -> one ds_store_b128 per thread.
// ---------------------------------------------------------------------------
__global__ void __launch_bounds__(256)
k_gemm_wmma(const float* __restrict__ A, const float* __restrict__ Bw,
            const float* __restrict__ bias, float* __restrict__ C,
            int M, int N, int K, int lda, int ldb, int ldc,
            long sA1, long sA2, long sB1, long sB2, long sC1, long sC2,
            int bmod, int act, int accum)
{
    const int z  = blockIdx.z;
    const int z1 = z / bmod, z2 = z % bmod;
    A  += (long)z1 * sA1 + (long)z2 * sA2;
    Bw += (long)z1 * sB1 + (long)z2 * sB2;
    C  += (long)z1 * sC1 + (long)z2 * sC2;

    const int bm   = blockIdx.y * 64;
    const int bn   = blockIdx.x * 64;
    const int tid  = threadIdx.x;
    const int wv   = tid >> 5;
    const int lane = tid & 31;
    const int hf   = lane >> 4;      // 0: lanes 0-15, 1: lanes 16-31
    const int r    = lane & 15;
    const int tn   = wv >> 1;        // 0..3 -> N tile
    const int m0   = (wv & 1) * 32;  // wave owns M rows m0..m0+31 (2 tiles)

    __shared__ alignas(16) _Float16 As[2][64][40];
    __shared__ alignas(16) _Float16 Bs[2][64][40];

    const int srow = tid >> 2;       // 0..63: staging row
    const int soff = (tid & 3) * 8;  // 0,8,16,24: staging col (8 floats/thread)

    auto stage = [&](int bsel, int k0) {
        // ---------------- A tile 64x32 ----------------
        if (bm + 64 <= M && k0 + 32 <= K) {
            const float* src = A + (long)(bm + srow) * lda + k0 + soff;
            float2 f0 = *(const float2*)(src + 0);
            float2 f1 = *(const float2*)(src + 2);
            float2 f2 = *(const float2*)(src + 4);
            float2 f3 = *(const float2*)(src + 6);
            if (k0 + 64 < K) __builtin_prefetch(src + 64, 0, 1);   // global_prefetch_b8
            v8h hv;
            hv[0] = (_Float16)f0.x; hv[1] = (_Float16)f0.y;
            hv[2] = (_Float16)f1.x; hv[3] = (_Float16)f1.y;
            hv[4] = (_Float16)f2.x; hv[5] = (_Float16)f2.y;
            hv[6] = (_Float16)f3.x; hv[7] = (_Float16)f3.y;
            *(v8h*)&As[bsel][srow][soff] = hv;
        } else {
            #pragma unroll
            for (int i = 0; i < 8; ++i) {
                int e  = tid + i * 256;
                int rr = e >> 5, kk = e & 31;
                int gm = bm + rr, gk = k0 + kk;
                int gmc = gm < M ? gm : M - 1;     // clamp: keep load unconditional
                int gkc = gk < K ? gk : K - 1;
                float v = A[(long)gmc * lda + gkc];
                As[bsel][rr][kk] = (_Float16)((gm < M && gk < K) ? v : 0.f);
            }
        }
        // ---------------- B tile 64x32 (Bw is [N,K] row-major == B^T) --------
        if (bn + 64 <= N && k0 + 32 <= K) {
            const float* src = Bw + (long)(bn + srow) * ldb + k0 + soff;
            float2 f0 = *(const float2*)(src + 0);
            float2 f1 = *(const float2*)(src + 2);
            float2 f2 = *(const float2*)(src + 4);
            float2 f3 = *(const float2*)(src + 6);
            if (k0 + 64 < K) __builtin_prefetch(src + 64, 0, 1);
            v8h hv;
            hv[0] = (_Float16)f0.x; hv[1] = (_Float16)f0.y;
            hv[2] = (_Float16)f1.x; hv[3] = (_Float16)f1.y;
            hv[4] = (_Float16)f2.x; hv[5] = (_Float16)f2.y;
            hv[6] = (_Float16)f3.x; hv[7] = (_Float16)f3.y;
            *(v8h*)&Bs[bsel][srow][soff] = hv;
        } else {
            #pragma unroll
            for (int i = 0; i < 8; ++i) {
                int e  = tid + i * 256;
                int rr = e >> 5, kk = e & 31;
                int gn = bn + rr, gk = k0 + kk;
                int gnc = gn < N ? gn : N - 1;
                int gkc = gk < K ? gk : K - 1;
                float v = Bw[(long)gnc * ldb + gkc];
                Bs[bsel][rr][kk] = (_Float16)((gn < N && gk < K) ? v : 0.f);
            }
        }
    };

    v8f acc0 = {};
    v8f acc1 = {};
    const int nk = (K + 31) >> 5;

    stage(0, 0);
    __syncthreads();
    int buf = 0;
    for (int kt = 0; kt < nk; ++kt) {
        if (kt + 1 < nk) stage(buf ^ 1, (kt + 1) << 5);   // overlap next tile with WMMA

        // A fragment (16x32, 16-bit): lane row = M%16, halves = {kh..kh+7, kh+16..kh+23}
        v8h a0lo = *(const v8h*)&As[buf][m0 + r][hf * 8];
        v8h a0hi = *(const v8h*)&As[buf][m0 + r][hf * 8 + 16];
        v8h a1lo = *(const v8h*)&As[buf][m0 + 16 + r][hf * 8];
        v8h a1hi = *(const v8h*)&As[buf][m0 + 16 + r][hf * 8 + 16];
        // B fragment (32x16): lane col = N%16, halves = {kh..kh+7, kh+16..kh+23} (kh = 8*hf)
        v8h blo  = *(const v8h*)&Bs[buf][tn * 16 + r][hf * 8];
        v8h bhi  = *(const v8h*)&Bs[buf][tn * 16 + r][hf * 8 + 16];
        v16h av0 = __builtin_shufflevector(a0lo, a0hi, 0,1,2,3,4,5,6,7,8,9,10,11,12,13,14,15);
        v16h av1 = __builtin_shufflevector(a1lo, a1hi, 0,1,2,3,4,5,6,7,8,9,10,11,12,13,14,15);
        v16h bv  = __builtin_shufflevector(blo,  bhi,  0,1,2,3,4,5,6,7,8,9,10,11,12,13,14,15);

        acc0 = __builtin_amdgcn_wmma_f32_16x16x32_f16(false, av0, false, bv, (short)0, acc0, false, false);
        acc1 = __builtin_amdgcn_wmma_f32_16x16x32_f16(false, av1, false, bv, (short)0, acc1, false, false);
        __syncthreads();
        buf ^= 1;
    }

    // D layout: VGPR v -> m = v + 8*hf, n = lane%16
    #pragma unroll
    for (int vi = 0; vi < 8; ++vi) {
        int gn = bn + tn * 16 + r;
        if (gn >= N) continue;
        int gm = bm + m0 + hf * 8 + vi;
        if (gm < M) {
            float val = acc0[vi];
            if (accum) val += C[(long)gm * ldc + gn];
            if (bias)  val += bias[gn];
            if (act == ACT_RELU)       val = fmaxf(val, 0.f);
            else if (act == ACT_SIG)   val = sigm(val);
            else if (act == ACT_LRELU) val = (val > 0.f) ? val : 0.01f * val;
            C[(long)gm * ldc + gn] = val;
        }
        int gm1 = gm + 16;
        if (gm1 < M) {
            float val = acc1[vi];
            if (accum) val += C[(long)gm1 * ldc + gn];
            if (bias)  val += bias[gn];
            if (act == ACT_RELU)       val = fmaxf(val, 0.f);
            else if (act == ACT_SIG)   val = sigm(val);
            else if (act == ACT_LRELU) val = (val > 0.f) ? val : 0.01f * val;
            C[(long)gm1 * ldc + gn] = val;
        }
    }
}

// ---------------------------------------------------------------------------
// Small elementwise / helper kernels
// ---------------------------------------------------------------------------
__global__ void k_fill(float* p, long n)
{
    long i = (long)blockIdx.x * 256 + threadIdx.x;
    if (i < n) p[i] = 0.f;
}

// h1[N,B,W,64] = relu(x[b,w,n] * seq_W1[j] + seq_b1[j])
__global__ void k_h1(float* h1, const float* __restrict__ x,
                     const float* __restrict__ W1, const float* __restrict__ b1)
{
    long idx = (long)blockIdx.x * 256 + threadIdx.x;    // NBW*64
    if (idx >= (long)NBW * 64) return;
    int jj = idx & 63;
    long pos = idx >> 6;
    int n = pos / (BB * WW);
    int rem = pos - (long)n * (BB * WW);
    int b = rem / WW, w = rem % WW;
    float xv = x[((long)b * WW + w) * NN + n];
    h1[idx] = fmaxf(xv * W1[jj] + b1[jj], 0.f);
}

// GRU gate: gx/gh are [N,B,750] for current step; h,outs updated.
__global__ void k_gru_gate(const float* __restrict__ gx, const float* __restrict__ gh,
                           const float* __restrict__ bih, const float* __restrict__ bhh,
                           float* h, float* outs, int w)
{
    long idx = (long)blockIdx.x * 256 + threadIdx.x;    // N*B*250
    if (idx >= (long)NN * BB * NN) return;
    int j = idx % NN;
    long nb = idx / NN;
    int n = nb / BB;
    const float* gxp = gx + nb * H3;
    const float* ghp = gh + nb * H3;
    const float* bi  = bih + (long)n * H3;
    const float* bh  = bhh + (long)n * H3;
    float xr = gxp[j]        + bi[j],        hr = ghp[j]        + bh[j];
    float xz = gxp[NN + j]   + bi[NN + j],   hz = ghp[NN + j]   + bh[NN + j];
    float xn = gxp[2*NN + j] + bi[2*NN + j], hn = ghp[2*NN + j] + bh[2*NN + j];
    float rg = sigm(xr + hr);
    float zg = sigm(xz + hz);
    float ng = tanhf(xn + rg * hn);
    float hv = (1.f - zg) * ng + zg * h[idx];
    h[idx] = hv;
    outs[(nb * WW + w) * NN + j] = hv;
}

// scores + softmax over W + weighted sum + residual + LayerNorm -> wr[B,N,N]
__global__ void __launch_bounds__(256)
k_score_ln(const float* __restrict__ E1, const float* __restrict__ E2,
           const float* __restrict__ outs, const float* __restrict__ hT,
           const float* __restrict__ attn_v, const float* __restrict__ attn_b,
           const float* __restrict__ ln_g, const float* __restrict__ ln_b,
           float* wr)
{
    int nb = blockIdx.x;               // n*B + b
    int n = nb / BB, b = nb % BB;
    int tid = threadIdx.x;
    __shared__ float red[256];
    __shared__ float sc[WW];
    __shared__ float aw[WW];
    const float* e1 = E1 + (long)nb * NN;

    for (int w = 0; w < WW; ++w) {
        const float* e2 = E2 + ((long)nb * WW + w) * NN;
        float p = 0.f;
        for (int j = tid; j < NN; j += 256)
            p += attn_v[j] * tanhf(e1[j] + e2[j] + attn_b[j]);
        red[tid] = p; __syncthreads();
        for (int s = 128; s > 0; s >>= 1) { if (tid < s) red[tid] += red[tid + s]; __syncthreads(); }
        if (tid == 0) sc[w] = red[0];
        __syncthreads();
    }
    if (tid == 0) {
        float mx = sc[0];
        for (int w = 1; w < WW; ++w) mx = fmaxf(mx, sc[w]);
        float sum = 0.f;
        for (int w = 0; w < WW; ++w) { float e = __expf(sc[w] - mx); aw[w] = e; sum += e; }
        for (int w = 0; w < WW; ++w) aw[w] /= sum;
    }
    __syncthreads();

    int j = tid;
    float val = 0.f;
    if (j < NN) {
        const float* o = outs + (long)nb * WW * NN;
        float a = 0.f;
        for (int w = 0; w < WW; ++w) a += aw[w] * o[w * NN + j];
        val = a + hT[(long)nb * NN + j];
    }
    red[tid] = (j < NN) ? val : 0.f; __syncthreads();
    for (int s = 128; s > 0; s >>= 1) { if (tid < s) red[tid] += red[tid + s]; __syncthreads(); }
    float mu = red[0] / (float)NN; __syncthreads();
    float d = (j < NN) ? (val - mu) : 0.f;
    red[tid] = d * d; __syncthreads();
    for (int s = 128; s > 0; s >>= 1) { if (tid < s) red[tid] += red[tid + s]; __syncthreads(); }
    float inv = rsqrtf(red[0] / (float)NN + 1e-5f);
    if (j < NN)
        wr[((long)b * NN + n) * NN + j] = (val - mu) * inv * ln_g[j] + ln_b[j];
}

// attention[N,N] = mean over (b,h) of softmax_k(q.k / sqrt(d))
__global__ void __launch_bounds__(256)
k_attn_pool(const float* __restrict__ q, const float* __restrict__ k, float* attention)
{
    int qn = blockIdx.x;
    int tid = threadIdx.x;
    __shared__ float logit[256];
    __shared__ float red[256];
    __shared__ float accum[256];
    accum[tid] = 0.f;
    const float scale = rsqrtf((float)DHEAD);
    __syncthreads();
    for (int b = 0; b < BB; ++b)
        for (int h = 0; h < HEADS; ++h) {
            float lg = -1e30f;
            if (tid < NN) {
                const float* qv = q + ((long)b * NN + qn) * NN + h * DHEAD;
                const float* kv = k + ((long)b * NN + tid) * NN + h * DHEAD;
                float d = 0.f;
                for (int dd = 0; dd < DHEAD; ++dd) d += qv[dd] * kv[dd];
                lg = d * scale;
            }
            logit[tid] = lg; red[tid] = lg; __syncthreads();
            for (int s = 128; s > 0; s >>= 1) { if (tid < s) red[tid] = fmaxf(red[tid], red[tid + s]); __syncthreads(); }
            float mx = red[0]; __syncthreads();
            float e = (tid < NN) ? __expf(logit[tid] - mx) : 0.f;
            red[tid] = e; __syncthreads();
            for (int s = 128; s > 0; s >>= 1) { if (tid < s) red[tid] += red[tid + s]; __syncthreads(); }
            float sum = red[0]; __syncthreads();
            if (tid < NN) accum[tid] += e / sum;
            __syncthreads();
        }
    if (tid < NN) attention[(long)qn * NN + tid] = accum[tid] / (float)(BB * HEADS);
}

__global__ void k_degree(const float* __restrict__ att, float* deg)
{
    int i = blockIdx.x, tid = threadIdx.x;
    __shared__ float red[256];
    red[tid] = (tid < NN) ? att[(long)i * NN + tid] : 0.f; __syncthreads();
    for (int s = 128; s > 0; s >>= 1) { if (tid < s) red[tid] += red[tid + s]; __syncthreads(); }
    if (tid == 0) deg[i] = red[0];
}

__global__ void k_cheb(const float* __restrict__ att, const float* __restrict__ deg, float* mul_L)
{
    int idx = blockIdx.x * 256 + threadIdx.x;
    if (idx >= NN * NN) return;
    int i = idx / NN, j = idx % NN;
    float na = 0.5f * (att[idx] + att[(long)j * NN + i]);
    float dhi = 1.f / (sqrtf(deg[i]) + 1e-7f);
    float dhj = 1.f / (sqrtf(deg[j]) + 1e-7f);
    float diag = (i == j) ? deg[i] : 0.f;
    float lap = dhi * (diag - na) * dhj;
    float I = (i == j) ? 1.f : 0.f;
    float L2 = lap;
    float L3 = 2.f * lap * L2 - I;
    float L4 = 2.f * lap * L3 - L2;
    mul_L[idx]              = I;
    mul_L[NN*NN + idx]      = L2;
    mul_L[2*NN*NN + idx]    = L3;
    mul_L[3*NN*NN + idx]    = L4;
}

// grouped [b,k,n,t] <-> flat [b,n,k*T+t]
__global__ void k_regroup(float* dst, const float* __restrict__ src,
                          int Bn, int Kc, int Nn, int Tl, int g2f)
{
    long total = (long)Bn * Kc * Nn * Tl;
    long idx = (long)blockIdx.x * 256 + threadIdx.x;
    if (idx >= total) return;
    int t = idx % Tl; long r = idx / Tl;
    int n = r % Nn; r /= Nn;
    int k = r % Kc; int b = r / Kc;
    long gi = (((long)b * Kc + k) * Nn + n) * Tl + t;
    long fi = ((long)b * Nn + n) * (long)(Kc * Tl) + (long)k * Tl + t;
    if (g2f) dst[fi] = src[gi];
    else     dst[gi] = src[fi];
}

__global__ void k_xcT(float* XcT, const float* __restrict__ Xc)
{
    int idx = blockIdx.x * 256 + threadIdx.x;   // B*12*250
    if (idx >= BB * TT * NN) return;
    int m = idx % NN; int r = idx / NN;
    int t = r % TT; int b = r / TT;
    XcT[idx] = Xc[((long)b * NN + m) * TT + t];
}

// mode 0: a*b ; 1: a+b ; 2: sigmoid(a-b)
__global__ void k_ew2(float* o, const float* __restrict__ a, const float* __restrict__ b,
                      long n, int mode)
{
    long i = (long)blockIdx.x * 256 + threadIdx.x;
    if (i >= n) return;
    float x = a[i], y = b[i];
    if (mode == 0)      o[i] = x * y;
    else if (mode == 1) o[i] = x + y;
    else                o[i] = sigm(x - y);
}

// DFT/IDFT matrices + sb_weight transpose
__global__ void k_init(float* Cf, float* Sf, float* Ci, float* Si,
                       float* sbwT, const float* __restrict__ sbw)
{
    int idx = blockIdx.x * 256 + threadIdx.x;
    const float PI2 = 6.28318530717958647692f;
    if (idx < 144) {
        int f = idx / TT, t = idx % TT;
        float th = PI2 * (float)(f * t) / (float)TT;
        Cf[idx] = cosf(th);
        Sf[idx] = -sinf(th);
    } else if (idx < 144 + 3600) {
        int j = idx - 144; int t = j / TS5, f = j % TS5;
        float th = PI2 * (float)(f * t) / (float)TS5;
        Ci[t * TS5 + f] = cosf(th) / (float)TS5;
        Si[t * TS5 + f] = -sinf(th) / (float)TS5;
    } else if (idx < 144 + 3600 + 2 * TS5 * OCT) {
        int j = idx - 144 - 3600;      // j = (s*60+sout)*240 + (k*60+t)
        int kt = j % OCT; int r = j / OCT;
        int sout = r % TS5; int s = r / TS5;
        int k = kt / TS5, t = kt % TS5;
        sbwT[j] = sbw[(((long)s * 4 + k) * TS5 + t) * TS5 + sout];
    }
}

__global__ void k_out(float* out, const float* __restrict__ o)
{
    int idx = blockIdx.x * 256 + threadIdx.x;   // B*3*N
    if (idx >= BB * 3 * NN) return;
    int n = idx % NN; int r = idx / NN;
    int hz = r % 3; int b = r / 3;
    out[idx] = o[((long)b * NN + n) * 3 + hz];
}

// ---------------------------------------------------------------------------
// Host side
// ---------------------------------------------------------------------------
static inline void gemm(hipStream_t st, const float* A, const float* B, const float* bias, float* C,
                        int M, int N, int K, int lda, int ldb, int ldc,
                        long sA1, long sA2, long sB1, long sB2, long sC1, long sC2,
                        int bmod, int batch, int act, int accum)
{
    dim3 g((unsigned)((N + 63) / 64), (unsigned)((M + 63) / 64), (unsigned)batch);
    k_gemm_wmma<<<g, 256, 0, st>>>(A, B, bias, C, M, N, K, lda, ldb, ldc,
                                   sA1, sA2, sB1, sB2, sC1, sC2, bmod, act, accum);
}
static inline void regroup(hipStream_t st, float* dst, const float* src,
                           int Bn, int Kc, int Nn, int Tl, int g2f)
{
    long total = (long)Bn * Kc * Nn * Tl;
    k_regroup<<<dim3((unsigned)((total + 255) / 256)), 256, 0, st>>>(dst, src, Bn, Kc, Nn, Tl, g2f);
}
static inline void ew2(hipStream_t st, float* o, const float* a, const float* b, long n, int mode)
{
    k_ew2<<<dim3((unsigned)((n + 255) / 256)), 256, 0, st>>>(o, a, b, n, mode);
}

// workspace layout (float offsets)
static const size_t F_h1   = 0;
static const size_t F_xsup = F_h1   + (size_t)NBW * 64;       //  6,144,000
static const size_t F_gx   = F_xsup + (size_t)NBW * 128;      // 18,432,000
static const size_t F_gh   = F_gx   + (size_t)NN * BB * H3;   // 24,432,000
static const size_t F_h    = F_gh   + (size_t)NN * BB * H3;   // 30,432,000
static const size_t F_outs = F_h    + (size_t)NN * BB * NN;   // 32,432,000
static const size_t F_E1   = F_outs + (size_t)NBW * NN;       // 56,432,000
static const size_t F_wr   = F_E1   + (size_t)NN * BB * NN;   // 58,432,000
static const size_t F_E2   = 0;                               // overlays h1..gh (free after GRU)
// stage-2 overlay (everything below F_h is dead after the score kernel)
static const size_t F_q    = 0;
static const size_t F_k2   = F_q    + (size_t)NB * NN;
static const size_t F_att  = F_k2   + (size_t)NB * NN;
static const size_t F_deg  = F_att  + NN * NN;
static const size_t F_mulL = F_deg  + 256;
static const size_t F_Xc   = F_mulL + 4 * NN * NN;
static const size_t F_XcT  = F_Xc   + NB * TT;
static const size_t F_gft  = F_XcT  + BB * TT * NN;
static const size_t F_ffr  = F_gft  + (size_t)BB * 4 * NN * TT;
static const size_t F_ffi  = F_ffr  + (size_t)BB * 4 * NN * TT;
static const size_t F_rT   = F_ffi  + (size_t)BB * 4 * NN * TT;
static const size_t F_iT   = F_rT   + (size_t)NB * 48;
static const size_t F_Lb   = F_iT   + (size_t)NB * 48;
static const size_t F_Rb   = F_Lb   + (size_t)NB * OCT;
static const size_t F_rA   = F_Rb   + (size_t)NB * OCT;
static const size_t F_iA   = F_rA   + (size_t)NB * OCT;
static const size_t F_rB   = F_iA   + (size_t)NB * OCT;
static const size_t F_iB   = F_rB   + (size_t)NB * OCT;
static const size_t F_gr   = F_iB   + (size_t)NB * OCT;
static const size_t F_gi   = F_gr   + (size_t)NB * OCT;
static const size_t F_gc   = F_gi   + (size_t)NB * OCT;
static const size_t F_gcp  = F_gc   + (size_t)NB * OCT;
static const size_t F_ig   = F_gcp  + (size_t)NB * OCT;
static const size_t F_fsrc = F_ig   + (size_t)NB * TS5;
static const size_t F_r0   = F_fsrc + (size_t)NB * TS5;
static const size_t F_r1   = F_r0   + NB * TT;
static const size_t F_bs   = F_r1   + NB * TT;
static const size_t F_bk   = F_bs   + NB * TT;
static const size_t F_fc   = F_bk   + NB * TT;
static const size_t F_hfc  = F_fc   + NB * TT;
static const size_t F_o    = F_hfc  + NB * TT;
// constants live ABOVE everything (never overlapped)
static const size_t F_Cf   = F_wr + (size_t)BB * NN * NN;     // 60,432,000
static const size_t F_Sf   = F_Cf + 144;
static const size_t F_Ci   = F_Sf + 144;
static const size_t F_Si   = F_Ci + 3600;
static const size_t F_sbwT = F_Si + 3600;                     // + 28,800

extern "C" void kernel_launch(void* const* d_in, const int* in_sizes, int n_in,
                              void* d_out, int out_size, void* d_ws, size_t ws_size,
                              hipStream_t stream)
{
    const float* x        = (const float*)d_in[0];
    const float* seq_W1   = (const float*)d_in[1];
    const float* seq_b1   = (const float*)d_in[2];
    const float* seq_W2   = (const float*)d_in[3];
    const float* seq_b2   = (const float*)d_in[4];
    const float* gru_Wih  = (const float*)d_in[5];
    const float* gru_Whh  = (const float*)d_in[6];
    const float* gru_bih  = (const float*)d_in[7];
    const float* gru_bhh  = (const float*)d_in[8];
    const float* attn_W   = (const float*)d_in[9];
    const float* attn_b   = (const float*)d_in[10];
    const float* attn_v   = (const float*)d_in[11];
    const float* ln_g     = (const float*)d_in[12];
    const float* ln_bv    = (const float*)d_in[13];
    const float* mha_Wq   = (const float*)d_in[14];
    const float* mha_bq   = (const float*)d_in[15];
    const float* mha_Wk   = (const float*)d_in[16];
    const float* mha_bk   = (const float*)d_in[17];
    const float* fc_ta_W  = (const float*)d_in[18];
    const float* fc_ta_b  = (const float*)d_in[19];
    const float* sb_w     = (const float*)d_in[20];
    const float* glu0_Wl  = (const float*)d_in[21];
    const float* glu0_bl  = (const float*)d_in[22];
    const float* glu0_Wr  = (const float*)d_in[23];
    const float* glu0_br  = (const float*)d_in[24];
    const float* g12_Wl   = (const float*)d_in[25];
    const float* g12_bl   = (const float*)d_in[26];
    const float* g12_Wr   = (const float*)d_in[27];
    const float* g12_br   = (const float*)d_in[28];
    const float* fore_W   = (const float*)d_in[29];
    const float* fore_b   = (const float*)d_in[30];
    const float* fres_W   = (const float*)d_in[31];
    const float* fres_b   = (const float*)d_in[32];
    const float* back_W   = (const float*)d_in[33];
    const float* back_b   = (const float*)d_in[34];
    const float* bs_W     = (const float*)d_in[35];
    const float* bs_b     = (const float*)d_in[36];
    const float* fc1_W    = (const float*)d_in[37];
    const float* fc1_b    = (const float*)d_in[38];
    const float* fc2_W    = (const float*)d_in[39];
    const float* fc2_b    = (const float*)d_in[40];
    float* out = (float*)d_out;
    float* WS  = (float*)d_ws;
    (void)in_sizes; (void)n_in; (void)out_size; (void)ws_size;
#define P(off) (WS + (off))

    // constants + zero hidden state
    k_init<<<dim3((144 + 3600 + 2 * TS5 * OCT + 255) / 256), 256, 0, stream>>>(
        P(F_Cf), P(F_Sf), P(F_Ci), P(F_Si), P(F_sbwT), sb_w);
    k_fill<<<dim3(((long)NN * BB * NN + 255) / 256), 256, 0, stream>>>(P(F_h), (long)NN * BB * NN);

    // embedding MLP
    k_h1<<<dim3(((long)NBW * 64 + 255) / 256), 256, 0, stream>>>(P(F_h1), x, seq_W1, seq_b1);
    gemm(stream, P(F_h1), seq_W2, seq_b2, P(F_xsup), NBW, 128, 64, 64, 64, 128,
         0, 0, 0, 0, 0, 0, 1, 1, ACT_RELU, 0);

    // per-node GRU over W=12 steps (batched per-node WMMA GEMMs; Whh stays L2-resident)
    for (int w = 0; w < WW; ++w) {
        gemm(stream, P(F_xsup) + (long)w * 128, gru_Wih, nullptr, P(F_gx),
             BB, H3, 128, WW * 128, 128, H3,
             (long)BB * WW * 128, 0, (long)H3 * 128, 0, (long)BB * H3, 0,
             1, NN, ACT_NONE, 0);
        gemm(stream, P(F_h), gru_Whh, nullptr, P(F_gh),
             BB, H3, NN, NN, NN, H3,
             (long)BB * NN, 0, (long)H3 * NN, 0, (long)BB * H3, 0,
             1, NN, ACT_NONE, 0);
        k_gru_gate<<<dim3(((long)NN * BB * NN + 255) / 256), 256, 0, stream>>>(
            P(F_gx), P(F_gh), gru_bih, gru_bhh, P(F_h), P(F_outs), w);
    }

    // additive time attention + LayerNorm -> wr[B,N,N]
    gemm(stream, P(F_h), attn_W, nullptr, P(F_E1), NB, NN, NN, NN, 2 * NN, NN,
         0, 0, 0, 0, 0, 0, 1, 1, ACT_NONE, 0);
    gemm(stream, P(F_outs), attn_W + NN, nullptr, P(F_E2), NBW, NN, NN, NN, 2 * NN, NN,
         0, 0, 0, 0, 0, 0, 1, 1, ACT_NONE, 0);
    k_score_ln<<<dim3(NB), 256, 0, stream>>>(P(F_E1), P(F_E2), P(F_outs), P(F_h),
                                             attn_v, attn_b, ln_g, ln_bv, P(F_wr));

    // self-attention -> pooled attention matrix -> Chebyshev basis
    gemm(stream, P(F_wr), mha_Wq, mha_bq, P(F_q), NB, NN, NN, NN, NN, NN,
         0, 0, 0, 0, 0, 0, 1, 1, ACT_NONE, 0);
    gemm(stream, P(F_wr), mha_Wk, mha_bk, P(F_k2), NB, NN, NN, NN, NN, NN,
         0, 0, 0, 0, 0, 0, 1, 1, ACT_NONE, 0);
    k_attn_pool<<<dim3(NN), 256, 0, stream>>>(P(F_q), P(F_k2), P(F_att));
    k_degree<<<dim3(NN), 256, 0, stream>>>(P(F_att), P(F_deg));
    k_cheb<<<dim3((NN * NN + 255) / 256), 256, 0, stream>>>(P(F_att), P(F_deg), P(F_mulL));

    // Xc = wr @ fc_ta^T + b
    gemm(stream, P(F_wr), fc_ta_W, fc_ta_b, P(F_Xc), NB, TT, NN, NN, NN, TT,
         0, 0, 0, 0, 0, 0, 1, 1, ACT_NONE, 0);
    k_xcT<<<dim3((BB * TT * NN + 255) / 256), 256, 0, stream>>>(P(F_XcT), P(F_Xc));

    auto glu = [&](const float* in, int Kin, const float* Wl, const float* bl,
                   const float* Wr2, const float* br2, float* ob) {
        gemm(stream, in, Wl,  bl,  P(F_Lb), NB, OCT, Kin, Kin, Kin, OCT,
             0, 0, 0, 0, 0, 0, 1, 1, ACT_NONE, 0);
        gemm(stream, in, Wr2, br2, P(F_Rb), NB, OCT, Kin, Kin, Kin, OCT,
             0, 0, 0, 0, 0, 0, 1, 1, ACT_SIG, 0);
        ew2(stream, ob, P(F_Lb), P(F_Rb), (long)NB * OCT, 0);
    };

    for (int s = 0; s < 2; ++s) {
        // gfted[b,k,n,t] = mul_L[k] @ Xc[b]   (batched: z1=b, z2=k)
        gemm(stream, P(F_mulL), P(F_XcT), nullptr, P(F_gft),
             NN, TT, NN, NN, NN, TT,
             0, (long)NN * NN, (long)TT * NN, 0, (long)4 * NN * TT, (long)NN * TT,
             4, BB * 4, ACT_NONE, 0);
        // FFT over T=12 as DFT matmuls
        gemm(stream, P(F_gft), P(F_Cf), nullptr, P(F_ffr), BB * 4 * NN, TT, TT, TT, TT, TT,
             0, 0, 0, 0, 0, 0, 1, 1, ACT_NONE, 0);
        gemm(stream, P(F_gft), P(F_Sf), nullptr, P(F_ffi), BB * 4 * NN, TT, TT, TT, TT, TT,
             0, 0, 0, 0, 0, 0, 1, 1, ACT_NONE, 0);
        regroup(stream, P(F_rT), P(F_ffr), BB, 4, NN, TT, 1);
        regroup(stream, P(F_iT), P(F_ffi), BB, 4, NN, TT, 1);
        // GLU0
        glu(P(F_rT), 48, glu0_Wl + (s * 2 + 0) * 11520, glu0_bl + (s * 2 + 0) * OCT,
            glu0_Wr + (s * 2 + 0) * 11520, glu0_br + (s * 2 + 0) * OCT, P(F_rA));
        glu(P(F_iT), 48, glu0_Wl + (s * 2 + 1) * 11520, glu0_bl + (s * 2 + 1) * OCT,
            glu0_Wr + (s * 2 + 1) * 11520, glu0_br + (s * 2 + 1) * OCT, P(F_iA));
        // GLU1, GLU2 (ping-pong)
        for (int i = 0; i < 2; ++i) {
            float* rin  = (i == 0) ? P(F_rA) : P(F_rB);
            float* rout = (i == 0) ? P(F_rB) : P(F_rA);
            float* iin  = (i == 0) ? P(F_iA) : P(F_iB);
            float* iout = (i == 0) ? P(F_iB) : P(F_iA);
            long wo0 = ((long)(s * 2 + i) * 2 + 0) * 57600, bo0 = ((long)(s * 2 + i) * 2 + 0) * OCT;
            long wo1 = ((long)(s * 2 + i) * 2 + 1) * 57600, bo1 = ((long)(s * 2 + i) * 2 + 1) * OCT;
            glu(rin, OCT, g12_Wl + wo0, g12_bl + bo0, g12_Wr + wo0, g12_br + bo0, rout);
            glu(iin, OCT, g12_Wl + wo1, g12_bl + bo1, g12_Wr + wo1, g12_br + bo1, iout);
        }
        // back to [b,k,n,60] and inverse DFT (real part)
        regroup(stream, P(F_gr), P(F_rA), BB, 4, NN, TS5, 0);
        regroup(stream, P(F_gi), P(F_iA), BB, 4, NN, TS5, 0);
        gemm(stream, P(F_gr), P(F_Ci), nullptr, P(F_gc), BB * 4 * NN, TS5, TS5, TS5, TS5, TS5,
             0, 0, 0, 0, 0, 0, 1, 1, ACT_NONE, 0);
        gemm(stream, P(F_gi), P(F_Si), nullptr, P(F_gc), BB * 4 * NN, TS5, TS5, TS5, TS5, TS5,
             0, 0, 0, 0, 0, 0, 1, 1, ACT_NONE, 1);
        regroup(stream, P(F_gcp), P(F_gc), BB, 4, NN, TS5, 1);
        // igfted = sum_k,t gconv * sb_weight
        gemm(stream, P(F_gcp), P(F_sbwT) + (long)s * TS5 * OCT, nullptr, P(F_ig),
             NB, TS5, OCT, OCT, OCT, TS5, 0, 0, 0, 0, 0, 0, 1, 1, ACT_NONE, 0);
        // forecast head
        gemm(stream, P(F_ig), fore_W + s * 3600, fore_b + s * TS5, P(F_fsrc),
             NB, TS5, TS5, TS5, TS5, TS5, 0, 0, 0, 0, 0, 0, 1, 1, ACT_SIG, 0);
        gemm(stream, P(F_fsrc), fres_W + s * TT * TS5, fres_b + s * TT,
             (s == 0) ? P(F_r0) : P(F_r1),
             NB, TT, TS5, TS5, TS5, TT, 0, 0, 0, 0, 0, 0, 1, 1, ACT_NONE, 0);
        if (s == 0) {
            gemm(stream, P(F_Xc), bs_W, bs_b, P(F_bs), NB, TT, TT, TT, TT, TT,
                 0, 0, 0, 0, 0, 0, 1, 1, ACT_NONE, 0);
            gemm(stream, P(F_ig), back_W, back_b, P(F_bk), NB, TT, TS5, TS5, TS5, TT,
                 0, 0, 0, 0, 0, 0, 1, 1, ACT_NONE, 0);
            ew2(stream, P(F_Xc), P(F_bk), P(F_bs), (long)NB * TT, 2);  // sigmoid(back - back_short)
            k_xcT<<<dim3((BB * TT * NN + 255) / 256), 256, 0, stream>>>(P(F_XcT), P(F_Xc));
        }
    }

    // forecast -> fc1 (leaky relu) -> fc2 -> transpose
    ew2(stream, P(F_fc), P(F_r0), P(F_r1), (long)NB * TT, 1);
    gemm(stream, P(F_fc), fc1_W, fc1_b, P(F_hfc), NB, TT, TT, TT, TT, TT,
         0, 0, 0, 0, 0, 0, 1, 1, ACT_LRELU, 0);
    gemm(stream, P(F_hfc), fc2_W, fc2_b, P(F_o), NB, 3, TT, TT, TT, 3,
         0, 0, 0, 0, 0, 0, 1, 1, ACT_NONE, 0);
    k_out<<<dim3((BB * 3 * NN + 255) / 256), 256, 0, stream>>>(out, P(F_o));
#undef P
}
